// InteractionBlock_12086037971135
// MI455X (gfx1250) — matrix operand order repaired and checked by reading, compiled, not run on hardware
//
#include <hip/hip_runtime.h>
#include <hip/hip_bf16.h>

// ---------------------------------------------------------------------------
// DimeNet++ InteractionBlock for MI455X (gfx1250, wave32, WMMA bf16->f32).
// All H x H GEMMs run on v_wmma_f32_16x16x32_bf16; activations kept in f32.
// Triplet kernel stages weights via TDM (tensor_load_to_lds) and gathers
// activation rows via async global->LDS loads when the builtins exist.
// ---------------------------------------------------------------------------

#define H      128
#define OUTD   128
#define E_N    50000
#define T_N    200000
#define NBIL   8
#define NRAD   6
#define NSBF   42      // NS*NR = 7*6
#define AP     136     // padded LDS row stride (bf16 elems) to dodge bank conflicts
#define MATSZ  16384   // 128*128 elements per weight matrix

#define AS3 __attribute__((address_space(3)))

#if defined(__gfx1250__) && __has_builtin(__builtin_amdgcn_global_load_async_to_lds_b128)
#define HAVE_ASYNC_B128 1
#else
#define HAVE_ASYNC_B128 0
#endif
#if defined(__gfx1250__) && __has_builtin(__builtin_amdgcn_tensor_load_to_lds)
#define HAVE_TDM 1
#else
#define HAVE_TDM 0
#endif

typedef __attribute__((ext_vector_type(16))) __bf16 v16bf;
typedef __attribute__((ext_vector_type(8)))  float  v8f;
typedef __attribute__((ext_vector_type(4)))  unsigned int v4u;
typedef __attribute__((ext_vector_type(8)))  int    v8i;
typedef __attribute__((ext_vector_type(4)))  int    v4i;

#define WMMA_BF16(A, B, C) \
  __builtin_amdgcn_wmma_f32_16x16x32_bf16(false, (A), false, (B), (short)0, (C), false, false)

__device__ __forceinline__ void wait_async0() {
#if defined(__gfx1250__)
#if __has_builtin(__builtin_amdgcn_s_wait_asynccnt)
  __builtin_amdgcn_s_wait_asynccnt(0);
#else
  asm volatile("s_wait_asynccnt 0" ::: "memory");
#endif
#endif
}

__device__ __forceinline__ void wait_tensor0() {
#if defined(__gfx1250__)
#if __has_builtin(__builtin_amdgcn_s_wait_tensorcnt)
  __builtin_amdgcn_s_wait_tensorcnt(0);
#else
  asm volatile("s_wait_tensorcnt 0" ::: "memory");
#endif
#endif
}

__device__ __forceinline__ float silu_f(float v) {
  return v / (1.0f + __expf(-v));
}

// A fragment: 16x32 bf16 tile, row m = lane&15, K pattern per ISA:
//   e<8  -> K = kk + half*8 + e
//   e>=8 -> K = kk + 16 + half*8 + (e-8)
__device__ __forceinline__ v16bf load_a_frag(const __bf16* __restrict__ arow, int kk, int half) {
  const __bf16* p0 = arow + kk + half * 8;
  v16bf a;
#pragma unroll
  for (int e = 0; e < 8; ++e) { a[e] = p0[e]; a[e + 8] = p0[16 + e]; }
  return a;
}

// B fragment from pre-packed layout frag[ks][n][e] = W[k = ks*16+e][n]:
// lane column n, element e<8 -> K = kk + half*8 + e; e>=8 -> K = kk + 16 + half*8 + (e-8)
__device__ __forceinline__ v16bf load_b_frag(const __bf16* __restrict__ base, int kk, int half, int n) {
  const __bf16* p0 = base + ((kk >> 4) * H + n) * 16 + half * 8;
  const __bf16* p1 = base + (((kk >> 4) + 1) * H + n) * 16 + half * 8;
  v16bf b;
#pragma unroll
  for (int e = 0; e < 8; ++e) { b[e] = p0[e]; b[e + 8] = p1[e]; }
  return b;
}

// One wave computes a 16x16 f32 tile of (A16x128 @ B128x16), K in 4 steps of 32.
__device__ __forceinline__ v8f wave_gemm128(const __bf16* __restrict__ Alds,
                                            const __bf16* __restrict__ Bfrag,
                                            int lane, int n) {
  const int m = lane & 15, half = lane >> 4;
  const __bf16* arow = Alds + m * AP;
  v8f acc = {};
#pragma unroll
  for (int ks = 0; ks < 4; ++ks) {
    v16bf a = load_a_frag(arow, ks * 32, half);
    v16bf b = load_b_frag(Bfrag, ks * 32, half, n);
    acc = WMMA_BF16(a, b, acc);
  }
  return acc;
}

// ---------------------------------------------------------------------------
// Weight conversion: f32 row-major W[in][out] -> bf16 B-fragment order.
// dst[((ks*128)+n)*16 + e] = W[(ks*16+e)*128 + n]
// ---------------------------------------------------------------------------
__global__ void __launch_bounds__(256) k_conv_mat(const float* __restrict__ W,
                                                  __bf16* __restrict__ dst) {
  int idx = blockIdx.x * 256 + threadIdx.x;      // 64 blocks -> 16384
  int ks = idx >> 11;
  int n  = (idx >> 4) & 127;
  int e  = idx & 15;
  dst[idx] = (__bf16)W[(ks * 16 + e) * H + n];
}

// Bilinear weight: Wbil[i][j][l] -> frag[j][ks][n=i][e], B_j[k=l][n=i] = Wbil[i][j][l]
__global__ void __launch_bounds__(256) k_conv_bil(const float* __restrict__ Wbil,
                                                  __bf16* __restrict__ dst) {
  int idx = blockIdx.x * 256 + threadIdx.x;      // 512 blocks -> 131072
  int j  = idx >> 14;
  int ks = (idx >> 11) & 7;
  int n  = (idx >> 4) & 127;
  int e  = idx & 15;
  dst[idx] = (__bf16)Wbil[(n * NBIL + j) * H + ks * 16 + e];
}

// ---------------------------------------------------------------------------
// Edge preprocessing: rbf_h = rbf@W_rbf; x_ji = silu(x@Wji+bji) (f32);
// x_kj = bf16( silu(x@Wkj+bkj) * rbf_h ).  One WG = 16 edges, 8 waves = 8 N-tiles.
// ---------------------------------------------------------------------------
__global__ void __launch_bounds__(256) k_edge_pre(
    const float* __restrict__ x, const float* __restrict__ rbf,
    const float* __restrict__ W_rbf,
    const float* __restrict__ bkj, const float* __restrict__ bji,
    const __bf16* __restrict__ fWji, const __bf16* __restrict__ fWkj,
    float* __restrict__ x_ji, __bf16* __restrict__ x_kj) {
  __shared__ __align__(16) __bf16 Ax[16 * AP];
  const int tid = threadIdx.x;
  const int e0  = blockIdx.x * 16;

  {  // stage x tile as bf16 A
    int m = tid >> 4, c = (tid & 15) * 8;
    const float* src = x + (size_t)(e0 + m) * H + c;
    __bf16* dst = Ax + m * AP + c;
#pragma unroll
    for (int q = 0; q < 8; ++q) dst[q] = (__bf16)src[q];
  }
  __syncthreads();

  const int wave = tid >> 5, lane = tid & 31;
  const int m = lane & 15, half = lane >> 4;
  const int n = wave * 16 + m;

  v8f accJ = {}, accK = {};
  const __bf16* arow = Ax + m * AP;
#pragma unroll
  for (int ks = 0; ks < 4; ++ks) {
    v16bf a  = load_a_frag(arow, ks * 32, half);
    v16bf bj = load_b_frag(fWji, ks * 32, half, n);
    v16bf bk = load_b_frag(fWkj, ks * 32, half, n);
    accJ = WMMA_BF16(a, bj, accJ);
    accK = WMMA_BF16(a, bk, accK);
  }

  const float bJ = bji[n], bK = bkj[n];
#pragma unroll
  for (int r = 0; r < 8; ++r) {
    int mr  = r + 8 * half;
    int row = e0 + mr;
    float rh = 0.f;
#pragma unroll
    for (int q = 0; q < NRAD; ++q) rh += rbf[row * NRAD + q] * W_rbf[q * H + n];
    float vj = silu_f(accJ[r] + bJ);
    float vk = silu_f(accK[r] + bK) * rh;
    x_ji[(size_t)row * H + n] = vj;
    x_kj[(size_t)row * H + n] = (__bf16)vk;
  }
}

// ---------------------------------------------------------------------------
// Triplet kernel: msg = sum_j sbf_h[:,j] * (xg @ Wbil_j^T); scatter-add by idx_ji.
// Persistent WGs: 256KB bilinear weight staged in LDS once via the Tensor Data
// Mover; per-tile activation gather uses async global->LDS loads (ASYNCcnt).
// ---------------------------------------------------------------------------
__global__ void __launch_bounds__(256) k_triplet(
    const float* __restrict__ sbf, const float* __restrict__ W_sbf,
    const int* __restrict__ idx_kj, const int* __restrict__ idx_ji,
    const __bf16* __restrict__ x_kj, const __bf16* __restrict__ fbil,
    float* __restrict__ agg) {
  __shared__ __align__(16) __bf16 Blds[NBIL * 8 * H * 16];  // 256 KB
  __shared__ __align__(16) __bf16 Xg[16 * AP];              // gathered x_kj rows
  __shared__ float Sh[16 * NBIL];
  __shared__ int   Ji[16];

  const int tid = threadIdx.x;
  const int wave = tid >> 5, lane = tid & 31;
  const int m = lane & 15, half = lane >> 4;
  const int n = wave * 16 + m;

#if HAVE_TDM
  if (wave == 0) {
    // 1-D TDM descriptor: copy 262144B (32768 x 8B elems) fbil -> Blds.
    unsigned long long ga = (unsigned long long)fbil;
    unsigned int lds_off = (unsigned int)(unsigned long long)(AS3 void*)Blds;
    v4u g0;
    g0[0] = 1u;                                   // count=1, user descriptor
    g0[1] = lds_off;                              // lds_addr (bits 63:32)
    g0[2] = (unsigned int)ga;                     // global_addr[31:0]
    g0[3] = (unsigned int)((ga >> 32) & 0x1FFFFFFu) | (2u << 30);  // addr[56:32] | type=2
    v8i g1;
    g1[0] = 3 << 16;                              // data_size = 8B
    g1[1] = (int)0x80000000;                      // tensor_dim0 = 32768 (low16 << 16)
    g1[2] = 0x00010000;                           // tensor_dim0 hi=0 | tensor_dim1 = 1
    g1[3] = (int)0x80000000;                      // tensor_dim1 hi=0 | tile_dim0 = 32768
    g1[4] = 0;                                    // tile_dim1/2 unused
    g1[5] = 32768;                                // tensor_dim0_stride
    g1[6] = 0;
    g1[7] = 0;
    v4i z4 = {0, 0, 0, 0};
    v8i z8 = {0, 0, 0, 0, 0, 0, 0, 0};
    __builtin_amdgcn_tensor_load_to_lds(g0, g1, z4, z4, z8, 0);
    wait_tensor0();
  }
#else
  {  // fallback: coalesced 16B-per-thread copy of the bilinear weights
    const uint4* src = (const uint4*)fbil;
    uint4* dst = (uint4*)Blds;
#pragma unroll 4
    for (int i = tid; i < (NBIL * 8 * H * 16 * 2) / 16; i += 256) dst[i] = src[i];
  }
#endif

  const int ntiles = T_N / 16;
  for (int tile = blockIdx.x; tile < ntiles; tile += gridDim.x) {
    const int t0 = tile * 16;
    __syncthreads();  // LDS reuse barrier (also publishes the one-time B stage)

    {  // gather 16 rows of x_kj (bf16, 256B each), 16B per thread
      int mm = tid >> 4, c = tid & 15;
      int srow = idx_kj[t0 + mm];
#if HAVE_ASYNC_B128
      v4i* g = (v4i*)(x_kj + (size_t)srow * H) + c;
      v4i* l = (v4i*)(Xg + mm * AP) + c;
      __builtin_amdgcn_global_load_async_to_lds_b128(g, l, 0, 0);
#else
      const uint4* s = (const uint4*)(x_kj + (size_t)srow * H) + c;
      *(uint4*)(Xg + mm * AP + c * 8) = *s;
#endif
    }
    if (tid < 16) {
      Ji[tid] = idx_ji[t0 + tid];
      if (tile + (int)gridDim.x < ntiles) {
        __builtin_prefetch(idx_kj + t0 + gridDim.x * 16 + tid, 0, 1);
        __builtin_prefetch(idx_ji + t0 + gridDim.x * 16 + tid, 0, 1);
      }
    }
    if (tid < 128) {  // sbf_h tile: [16 x 8] = sbf @ W_sbf (overlaps async gather)
      int mm = tid >> 3, j = tid & 7;
      const float* srow = sbf + (size_t)(t0 + mm) * NSBF;
      float acc = 0.f;
#pragma unroll
      for (int q = 0; q < NSBF; ++q) acc += srow[q] * W_sbf[q * NBIL + j];
      Sh[mm * NBIL + j] = acc;
    }
#if HAVE_ASYNC_B128
    wait_async0();
#endif
    __syncthreads();

    v8f acc = {};
    const __bf16* arow = Xg + m * AP;
#pragma unroll
    for (int j = 0; j < NBIL; ++j) {
      const float s = Sh[m * NBIL + j];           // per-lane row scale (A row = lane&15)
      const __bf16* bbase = Blds + j * (8 * H * 16);
#pragma unroll
      for (int ks = 0; ks < 4; ++ks) {
        const __bf16* p0 = arow + ks * 32 + half * 8;
        v16bf a;
#pragma unroll
        for (int e = 0; e < 8; ++e) {
          a[e]     = (__bf16)((float)p0[e] * s);
          a[e + 8] = (__bf16)((float)p0[16 + e] * s);
        }
        v16bf b = load_b_frag(bbase, ks * 32, half, n);
        acc = WMMA_BF16(a, b, acc);
      }
    }
#pragma unroll
    for (int r = 0; r < 8; ++r) {
      int mr = r + 8 * half;
      atomicAdd(&agg[(size_t)Ji[mr] * H + n], acc[r]);
    }
  }
}

// ---------------------------------------------------------------------------
// Fused post chain per 16-edge tile:
// h = x_ji + agg; h += silu(silu(h@W1+b1)@W2+b2)   (before x1)
// h = silu(h@Wlin+blin) + x; 2x after-residual; out = silu(h@Wout+bout)
// ---------------------------------------------------------------------------
__device__ __forceinline__ void mlp_block(__bf16* Hb, __bf16* Tb, float* Hf,
                                          const __bf16* W1, const float* b1,
                                          const __bf16* W2, const float* b2,
                                          int lane, int n) {
  const int half = lane >> 4;
  v8f t = wave_gemm128(Hb, W1, lane, n);
  const float c1 = b1[n];
#pragma unroll
  for (int r = 0; r < 8; ++r) {
    int mr = r + 8 * half;
    Tb[mr * AP + n] = (__bf16)silu_f(t[r] + c1);
  }
  __syncthreads();
  v8f u = wave_gemm128(Tb, W2, lane, n);
  const float c2 = b2[n];
#pragma unroll
  for (int r = 0; r < 8; ++r) {
    int mr = r + 8 * half;
    float hv = Hf[mr * H + n] + silu_f(u[r] + c2);
    Hf[mr * H + n] = hv;
    Hb[mr * AP + n] = (__bf16)hv;
  }
  __syncthreads();
}

__global__ void __launch_bounds__(256) k_edge_post(
    const float* __restrict__ x, const float* __restrict__ x_ji,
    const float* __restrict__ agg, const __bf16* __restrict__ fW,
    const float* __restrict__ bb1, const float* __restrict__ bb2,
    const float* __restrict__ blin,
    const float* __restrict__ ab1, const float* __restrict__ ab2,
    const float* __restrict__ bout, float* __restrict__ out) {
  __shared__ __align__(16) __bf16 Hb[16 * AP];
  __shared__ __align__(16) __bf16 Tb[16 * AP];
  __shared__ float Hf[16 * H];

  const int tid = threadIdx.x;
  const int e0  = blockIdx.x * 16;

  {  // h0 = x_ji + agg
    int m = tid >> 4, c = (tid & 15) * 8;
    size_t base = (size_t)(e0 + m) * H + c;
#pragma unroll
    for (int q = 0; q < 8; ++q) {
      float v = x_ji[base + q] + agg[base + q];
      Hf[m * H + c + q] = v;
      Hb[m * AP + c + q] = (__bf16)v;
    }
  }
  __syncthreads();

  const int wave = tid >> 5, lane = tid & 31;
  const int half = lane >> 4;
  const int n = wave * 16 + (lane & 15);

  const __bf16* WbW1 = fW + 2 * MATSZ;
  const __bf16* WbW2 = fW + 3 * MATSZ;
  const __bf16* WLin = fW + 4 * MATSZ;
  const __bf16* Wa10 = fW + 5 * MATSZ;
  const __bf16* Wa20 = fW + 6 * MATSZ;
  const __bf16* Wa11 = fW + 7 * MATSZ;
  const __bf16* Wa21 = fW + 8 * MATSZ;
  const __bf16* WOut = fW + 9 * MATSZ;

  // before residual block
  mlp_block(Hb, Tb, Hf, WbW1, bb1, WbW2, bb2, lane, n);

  // h = silu(h@Wlin + blin) + x
  {
    v8f u = wave_gemm128(Hb, WLin, lane, n);
    __syncthreads();  // all reads of Hb complete before overwrite
    const float cl = blin[n];
#pragma unroll
    for (int r = 0; r < 8; ++r) {
      int mr = r + 8 * half;
      float hv = silu_f(u[r] + cl) + x[(size_t)(e0 + mr) * H + n];
      Hf[mr * H + n] = hv;
      Hb[mr * AP + n] = (__bf16)hv;
    }
    __syncthreads();
  }

  // after residual blocks x2
  mlp_block(Hb, Tb, Hf, Wa10, ab1, Wa20, ab2, lane, n);
  mlp_block(Hb, Tb, Hf, Wa11, ab1 + H, Wa21, ab2 + H, lane, n);

  // out = silu(h @ Wout + bout)
  {
    v8f o = wave_gemm128(Hb, WOut, lane, n);
    const float co = bout[n];
#pragma unroll
    for (int r = 0; r < 8; ++r) {
      int mr = r + 8 * half;
      out[(size_t)(e0 + mr) * H + n] = silu_f(o[r] + co);
    }
  }
}

// ---------------------------------------------------------------------------
extern "C" void kernel_launch(void* const* d_in, const int* in_sizes, int n_in,
                              void* d_out, int out_size, void* d_ws, size_t ws_size,
                              hipStream_t stream) {
  const float* x     = (const float*)d_in[0];
  const float* rbf   = (const float*)d_in[1];
  const float* sbf   = (const float*)d_in[2];
  const int*   ikj   = (const int*)d_in[3];
  const int*   iji   = (const int*)d_in[4];
  const float* W_rbf = (const float*)d_in[5];
  const float* W_sbf = (const float*)d_in[6];
  const float* Wkj   = (const float*)d_in[7];
  const float* bkj   = (const float*)d_in[8];
  const float* Wji   = (const float*)d_in[9];
  const float* bji   = (const float*)d_in[10];
  const float* Wbil  = (const float*)d_in[11];
  const float* bW1   = (const float*)d_in[12];
  const float* bb1   = (const float*)d_in[13];
  const float* bW2   = (const float*)d_in[14];
  const float* bb2   = (const float*)d_in[15];
  const float* Wlin  = (const float*)d_in[16];
  const float* blin  = (const float*)d_in[17];
  const float* aW1   = (const float*)d_in[18];
  const float* ab1   = (const float*)d_in[19];
  const float* aW2   = (const float*)d_in[20];
  const float* ab2   = (const float*)d_in[21];
  const float* Wout  = (const float*)d_in[22];
  const float* bout  = (const float*)d_in[23];
  (void)in_sizes; (void)n_in; (void)out_size; (void)ws_size;

  char* ws = (char*)d_ws;
  __bf16* fW   = (__bf16*)ws;                                       // 10 matrices
  __bf16* fbil = (__bf16*)(ws + 10 * MATSZ * 2);                    // 8 matrices
  float*  xji  = (float*)(ws + 18 * MATSZ * 2);
  __bf16* xkj  = (__bf16*)((char*)xji + (size_t)E_N * H * 4);
  float*  agg  = (float*)((char*)xkj + (size_t)E_N * H * 2);

  // weight conversions (f32 -> bf16 fragment layout)
  const float* mats[10] = {Wji, Wkj, bW1, bW2, Wlin,
                           aW1, aW2, aW1 + MATSZ, aW2 + MATSZ, Wout};
  for (int i = 0; i < 10; ++i)
    k_conv_mat<<<64, 256, 0, stream>>>(mats[i], fW + (size_t)i * MATSZ);
  k_conv_bil<<<512, 256, 0, stream>>>(Wbil, fbil);

  (void)hipMemsetAsync(agg, 0, (size_t)E_N * H * sizeof(float), stream);

  k_edge_pre<<<E_N / 16, 256, 0, stream>>>(x, rbf, W_rbf, bkj, bji,
                                           fW, fW + MATSZ, xji, xkj);
  k_triplet<<<512, 256, 0, stream>>>(sbf, W_sbf, ikj, iji, xkj, fbil, agg);
  k_edge_post<<<E_N / 16, 256, 0, stream>>>(x, xji, agg, fW, bb1, bb2, blin,
                                            ab1, ab2, bout, (float*)d_out);
}
